// GlmDSAMoEGate_62895501082720
// MI455X (gfx1250) — matrix operand order repaired and checked by reading, not compiled, not run on hardware
//
#include <hip/hip_runtime.h>
#include <hip/hip_bf16.h>
#include <math.h>

// ---------------------------------------------------------------------------
// Fused MoE gate (GLM/DeepSeek group-limited top-k router) for gfx1250.
//
//   logits = x(T x 4096) @ W^T(4096 x 256)      <- v_wmma_f32_16x16x32_bf16
//   scores = sigmoid(logits); choice = scores + bias
//   group top-2 sums -> top-4 groups -> masked top-8 -> norm * 2.5
//
// GEMM precision: bf16x3 split (hi*hi + lo*hi + hi*lo) with f32 accumulate,
// recovering ~fp32 accuracy on the fast bf16 WMMA path. Kernel is HBM-bound
// (256 MB of x @ 23.3 TB/s ~= 11 us), so the 3x matrix-op cost is hidden.
//
// Workgroup = 256 threads (8 wave32) = 16 tokens x 256 experts.
//   Phase 1: wave w computes expert tiles n = 2w, 2w+1 (16x16 f32 acc each).
//   Phase 2a: 256 thr  -> sigmoid+bias (1 token x 16 experts each)
//   Phase 2b: 128 thr  -> 16 tokens x 8 group top-2 sums
//   Phase 2c:  16 thr  -> top-4 groups + masked top-8 + normalize (serial,
//              rolled scans to keep static code small / I$-friendly)
// ---------------------------------------------------------------------------

typedef __attribute__((ext_vector_type(16))) __bf16 v16bf;
typedef __attribute__((ext_vector_type(8)))  float  v8f;

#define HDIM     4096
#define NEXP     256
#define TOPK     8
#define NGROUP   8
#define GSIZE    32      // NEXP / NGROUP
#define TOPKG    4
#define RSCALE   2.5f
#define MBLK     16      // tokens per workgroup

// Split 16 fp32 values into bf16 hi + bf16 residual lo.
__device__ __forceinline__ void split_bf16x16(const float* __restrict__ f,
                                              v16bf& hi, v16bf& lo) {
#pragma unroll
  for (int i = 0; i < 16; ++i) {
    __bf16 h = (__bf16)f[i];
    hi[i] = h;
    lo[i] = (__bf16)(f[i] - (float)h);
  }
}

__device__ __forceinline__ v8f wmma_bf16(v16bf a, v16bf b, v8f c) {
  // 8 args: (neg_a, A, neg_b, B, c_mod, C, reuse_a, reuse_b)
  return __builtin_amdgcn_wmma_f32_16x16x32_bf16(false, a, false, b,
                                                 (short)0, c, false, false);
}

__global__ __launch_bounds__(256)
void moe_gate_fused(const float* __restrict__ x,      // (T, 4096)
                    const float* __restrict__ w,      // (256, 4096)
                    const float* __restrict__ bias,   // (256,)
                    float* __restrict__ out,          // [T*8 idx][T*8 wts]
                    int T) {
  __shared__ float s_sc[MBLK][NEXP];     // logits, then sigmoid+bias
  __shared__ float s_gs[MBLK][NGROUP];   // group scores

  const int tid   = threadIdx.x;
  const int lane  = tid & 31;
  const int wave  = tid >> 5;
  const int half  = lane >> 4;           // K-half selector (contiguous K order here)
  const int l16   = lane & 15;           // M (for A) / N (for B,C)
  const int mbase = blockIdx.x * MBLK;

  // --- Phase 1: 16x256 logit tile via bf16x3 WMMA --------------------------
  int row = mbase + l16;
  if (row >= T) row = T - 1;             // clamp: keeps EXEC all-ones for WMMA
  // A (bf16 16x32), contiguous K order used here: lane half h holds K = h*16+[0..16)
  const float* xs = x + (size_t)row * HDIM + half * 16;

  const int n0 = (wave * 2 + 0) * 16;
  const int n1 = (wave * 2 + 1) * 16;
  // B (bf16 32x16), same contiguous K order as A: lane = column N, half h holds K = h*16+[0..16)
  const float* wr0 = w + (size_t)(n0 + l16) * HDIM + half * 16;
  const float* wr1 = w + (size_t)(n1 + l16) * HDIM + half * 16;

  v8f acc0 = {}; v8f acc1 = {};

#pragma unroll 1
  for (int kc = 0; kc < HDIM; kc += 32) {
    __builtin_prefetch(xs + kc + 64, 0, 1);   // global_prefetch_b8
    float af[16] __attribute__((aligned(16)));
    *(float4*)(af +  0) = *(const float4*)(xs + kc);
    *(float4*)(af +  4) = *(const float4*)(xs + kc + 4);
    *(float4*)(af +  8) = *(const float4*)(xs + kc + 8);
    *(float4*)(af + 12) = *(const float4*)(xs + kc + 12);
    v16bf a_hi, a_lo;
    split_bf16x16(af, a_hi, a_lo);

    float bf0[16] __attribute__((aligned(16)));
    *(float4*)(bf0 +  0) = *(const float4*)(wr0 + kc);
    *(float4*)(bf0 +  4) = *(const float4*)(wr0 + kc + 4);
    *(float4*)(bf0 +  8) = *(const float4*)(wr0 + kc + 8);
    *(float4*)(bf0 + 12) = *(const float4*)(wr0 + kc + 12);
    v16bf b_hi, b_lo;
    split_bf16x16(bf0, b_hi, b_lo);
    acc0 = wmma_bf16(a_hi, b_hi, acc0);
    acc0 = wmma_bf16(a_lo, b_hi, acc0);
    acc0 = wmma_bf16(a_hi, b_lo, acc0);

    float bf1[16] __attribute__((aligned(16)));
    *(float4*)(bf1 +  0) = *(const float4*)(wr1 + kc);
    *(float4*)(bf1 +  4) = *(const float4*)(wr1 + kc + 4);
    *(float4*)(bf1 +  8) = *(const float4*)(wr1 + kc + 8);
    *(float4*)(bf1 + 12) = *(const float4*)(wr1 + kc + 12);
    split_bf16x16(bf1, b_hi, b_lo);
    acc1 = wmma_bf16(a_hi, b_hi, acc1);
    acc1 = wmma_bf16(a_lo, b_hi, acc1);
    acc1 = wmma_bf16(a_hi, b_lo, acc1);
  }

  // C layout (32-bit C 16x16): VGPR i -> M = i + 8*half, N = l16
#pragma unroll
  for (int i = 0; i < 8; ++i) {
    s_sc[i + 8 * half][n0 + l16] = acc0[i];
    s_sc[i + 8 * half][n1 + l16] = acc1[i];
  }
  __syncthreads();

  // --- Phase 2a: sigmoid + bias, all 256 threads ---------------------------
  // thread -> token tid>>4, experts (tid&15)*16 .. +16
  {
    const int t  = tid >> 4;
    const int e0 = (tid & 15) * 16;
#pragma unroll
    for (int i = 0; i < 16; ++i) {
      const int e = e0 + i;
      float s = 1.0f / (1.0f + expf(-s_sc[t][e]));
      s_sc[t][e] = s + bias[e];          // choice; sigmoid = choice - bias
    }
  }
  __syncthreads();

  // --- Phase 2b: group scores (sum of top-2 per group), 128 threads --------
  if (tid < MBLK * NGROUP) {
    const int t = tid >> 3;
    const int g = tid & 7;
    float m1 = -INFINITY, m2 = -INFINITY;
#pragma unroll 1
    for (int j = 0; j < GSIZE; ++j) {
      float v = s_sc[t][g * GSIZE + j];
      if (v > m1)      { m2 = m1; m1 = v; }
      else if (v > m2) { m2 = v; }
    }
    s_gs[t][g] = m1 + m2;
  }
  __syncthreads();

  // --- Phase 2c: top-4 groups + masked top-8, 16 threads -------------------
  if (tid < MBLK) {
    const int t   = tid;
    const int tok = mbase + t;
    if (tok < T) {
      const float* sc = s_sc[t];
      // top-4 groups (strict >, ascending scan == lowest-index tie-break)
      unsigned gmask = 0;
#pragma unroll
      for (int k = 0; k < TOPKG; ++k) {
        float best = -INFINITY; int bi = 0;
#pragma unroll
        for (int g = 0; g < NGROUP; ++g)
          if (!((gmask >> g) & 1u) && s_gs[t][g] > best) {
            best = s_gs[t][g]; bi = g;
          }
        gmask |= 1u << bi;
      }
      // top-8 experts over masked scores (masked groups contribute 0.0,
      // matching jnp.where(mask>0, choice, 0.0)). k-loop unrolled so the
      // duplicate check against idx[0..k) is constant-indexed registers.
      int   idx[TOPK];
      float tw[TOPK];
      float sum = 0.0f;
#pragma unroll
      for (int k = 0; k < TOPK; ++k) {
        float best = -INFINITY; int bi = 0;
#pragma unroll 1
        for (int e = 0; e < NEXP; ++e) {
          float v = ((gmask >> (e >> 5)) & 1u) ? sc[e] : 0.0f;
          bool dup = false;
#pragma unroll
          for (int j = 0; j < TOPK; ++j)
            if (j < k && idx[j] == e) dup = true;
          if (!dup && v > best) { best = v; bi = e; }
        }
        idx[k] = bi;
        float s = sc[bi] - bias[bi];     // pure sigmoid score
        tw[k]  = s;
        sum   += s;
      }
      const float inv = RSCALE / (sum + 1e-20f);
      float* out_idx = out;
      float* out_wts = out + (size_t)T * TOPK;
#pragma unroll
      for (int k = 0; k < TOPK; ++k) {
        out_idx[(size_t)tok * TOPK + k] = (float)idx[k];
        out_wts[(size_t)tok * TOPK + k] = tw[k] * inv;
      }
    }
  }
}

extern "C" void kernel_launch(void* const* d_in, const int* in_sizes, int n_in,
                              void* d_out, int out_size, void* d_ws, size_t ws_size,
                              hipStream_t stream) {
  const float* x    = (const float*)d_in[0];   // hidden_states (4,4096,4096) f32
  const float* w    = (const float*)d_in[1];   // weight (256,4096) f32
  const float* bias = (const float*)d_in[2];   // e_score_correction_bias (256,) f32
  float* out = (float*)d_out;                  // [T*8 indices][T*8 weights]

  const int T = in_sizes[0] / HDIM;            // 16384 tokens
  const int grid = (T + MBLK - 1) / MBLK;      // 1024 workgroups
  moe_gate_fused<<<grid, 256, 0, stream>>>(x, w, bias, out, T);
}
